// StageBlocks_10101763080206
// MI455X (gfx1250) — matrix operand order repaired and checked by reading, not compiled, not run on hardware
//
#include <hip/hip_runtime.h>
#include <hip/hip_bf16.h>

// ---------------------------------------------------------------------------
// PointNeXt-style stage for MI455X (gfx1250, wave32, WMMA).
// ---------------------------------------------------------------------------

#define B_   4
#define N_   8192
#define CIN  64
#define CO   128
#define M_   2048
#define K_   32
#define L_   2
#define EPS_ 1e-5f

typedef __attribute__((ext_vector_type(16))) _Float16 v16h;
typedef __attribute__((ext_vector_type(8)))  float    v8f;

// ---------------------------------------------------------------------------
// Zero a float region (stats accumulators must be zeroed every call).
// ---------------------------------------------------------------------------
__global__ void zero_kernel(float* __restrict__ p, int n) {
  int i = blockIdx.x * blockDim.x + threadIdx.x;
  if (i < n) p[i] = 0.0f;
}

// ---------------------------------------------------------------------------
// Farthest point sampling: one workgroup per batch. dis[] lives in LDS.
// Writes centroids (B,M,3) directly (centroids == first tuple output).
// ---------------------------------------------------------------------------
__global__ __launch_bounds__(1024)
void fps_kernel(const float* __restrict__ pos, float* __restrict__ centroids) {
  __shared__ float dis[N_];
  __shared__ float rv[32];
  __shared__ int   ri[32];
  __shared__ int   s_last;
  const int b = blockIdx.x;
  const int t = threadIdx.x;
  const float* P = pos + (size_t)b * N_ * 3;
  float* C = centroids + (size_t)b * M_ * 3;

  for (int j = t; j < N_; j += 1024) dis[j] = 1e10f;
  if (t == 0) s_last = 0;
  __syncthreads();

  for (int s = 0; s < M_; ++s) {
    const int last = s_last;
    const float lx = P[last*3+0], ly = P[last*3+1], lz = P[last*3+2];
    if (t == 0) { C[s*3+0] = lx; C[s*3+1] = ly; C[s*3+2] = lz; }

    float bv = -1.0f; int bj = 0;
    for (int j = t; j < N_; j += 1024) {
      float dx = P[j*3+0]-lx, dy = P[j*3+1]-ly, dz = P[j*3+2]-lz;
      float d  = dx*dx + dy*dy + dz*dz;
      float nd = fminf(dis[j], d);
      dis[j] = nd;
      if (nd > bv) { bv = nd; bj = j; }
    }
    // wave32 reduce (max value, ties -> smaller index)
    for (int off = 16; off > 0; off >>= 1) {
      float ov = __shfl_down(bv, off, 32);
      int   oj = __shfl_down(bj, off, 32);
      if (ov > bv || (ov == bv && oj < bj)) { bv = ov; bj = oj; }
    }
    if ((t & 31) == 0) { rv[t >> 5] = bv; ri[t >> 5] = bj; }
    __syncthreads();
    if (t < 32) {
      bv = rv[t]; bj = ri[t];
      for (int off = 16; off > 0; off >>= 1) {
        float ov = __shfl_down(bv, off, 32);
        int   oj = __shfl_down(bj, off, 32);
        if (ov > bv || (ov == bv && oj < bj)) { bv = ov; bj = oj; }
      }
      if (t == 0) s_last = bj;
    }
    __syncthreads();
  }
}

// ---------------------------------------------------------------------------
// Ball query: one wave per (b, centroid). First K_ in-ball indices in index
// order (ballot + popc prefix), padded with the first hit (n-1 if none).
// ---------------------------------------------------------------------------
__global__ __launch_bounds__(256)
void ballq_kernel(const float* __restrict__ pts, const float* __restrict__ ctr,
                  int* __restrict__ sel, int n, float r2) {
  const int wid  = (blockIdx.x * blockDim.x + threadIdx.x) >> 5; // b*M_ + m
  const int lane = threadIdx.x & 31;
  const int b = wid / M_;
  const float* P = pts + (size_t)b * n * 3;
  const float cx = ctr[(size_t)wid*3+0];
  const float cy = ctr[(size_t)wid*3+1];
  const float cz = ctr[(size_t)wid*3+2];
  int* S = sel + (size_t)wid * K_;

  int cnt = 0, first = n - 1;
  for (int base = 0; base < n && cnt < K_; base += 32) {
    const int j = base + lane;
    float dx = P[j*3+0]-cx, dy = P[j*3+1]-cy, dz = P[j*3+2]-cz;
    bool  in = (dx*dx + dy*dy + dz*dz) < r2;
    unsigned msk = (unsigned)__ballot(in);
    if (cnt == 0 && msk != 0u) first = base + __builtin_ctz(msk);
    int pre = cnt + __popc(msk & ((1u << lane) - 1u));
    if (in && pre < K_) S[pre] = j;
    cnt += __popc(msk);
  }
  if (cnt > K_) cnt = K_;
  if (lane >= cnt) S[lane] = first;
}

// ---------------------------------------------------------------------------
// Gather normalized neighbor offsets gp=(p[sel]-c)*invR, store (b,m,k,3),
// accumulate global mu(3) + second moment S(6) -> 9 sums for analytic BN.
// ---------------------------------------------------------------------------
__global__ __launch_bounds__(256)
void gp_stats_kernel(const float* __restrict__ pts, const float* __restrict__ ctr,
                     const int* __restrict__ sel, float* __restrict__ gp3,
                     float* __restrict__ st, int n, float invR) {
  __shared__ float s9[9];
  const int t = threadIdx.x;
  if (t < 9) s9[t] = 0.0f;
  __syncthreads();

  const int i  = blockIdx.x * 256 + t;   // 0 .. B*M*K-1
  const int bm = i >> 5;                 // K_ == 32
  const int b  = bm / M_;
  const int idx = sel[i];
  const float* p = pts + ((size_t)b * n + idx) * 3;
  const float cx = ctr[(size_t)bm*3+0], cy = ctr[(size_t)bm*3+1], cz = ctr[(size_t)bm*3+2];
  const float gx = (p[0]-cx)*invR, gy = (p[1]-cy)*invR, gz = (p[2]-cz)*invR;
  gp3[(size_t)i*3+0] = gx; gp3[(size_t)i*3+1] = gy; gp3[(size_t)i*3+2] = gz;

  atomicAdd(&s9[0], gx);     atomicAdd(&s9[1], gy);     atomicAdd(&s9[2], gz);
  atomicAdd(&s9[3], gx*gx);  atomicAdd(&s9[4], gx*gy);  atomicAdd(&s9[5], gx*gz);
  atomicAdd(&s9[6], gy*gy);  atomicAdd(&s9[7], gy*gz);  atomicAdd(&s9[8], gz*gz);
  __syncthreads();
  if (t < 9) atomicAdd(&st[t], s9[t]);
}

// ---------------------------------------------------------------------------
// Fold training-mode BN of z = W3(128x3) * gp into per-channel (scale,bias)
// using linearity: mean_o = W*mu,  E[z^2]_o = (W S W^T)_oo.
// ---------------------------------------------------------------------------
__global__ void bn_fold3_kernel(const float* __restrict__ W3, const float* __restrict__ st,
                                float inv_cnt, const float* __restrict__ gamma,
                                const float* __restrict__ beta,
                                float* __restrict__ s_out, float* __restrict__ t_out) {
  const int o = threadIdx.x;   // 128 threads
  const float mu0 = st[0]*inv_cnt, mu1 = st[1]*inv_cnt, mu2 = st[2]*inv_cnt;
  const float Exx = st[3]*inv_cnt, Exy = st[4]*inv_cnt, Exz = st[5]*inv_cnt;
  const float Eyy = st[6]*inv_cnt, Eyz = st[7]*inv_cnt, Ezz = st[8]*inv_cnt;
  const float w0 = W3[o*3+0], w1 = W3[o*3+1], w2 = W3[o*3+2];
  const float mean = w0*mu0 + w1*mu1 + w2*mu2;
  const float ez2  = w0*w0*Exx + w1*w1*Eyy + w2*w2*Ezz
                   + 2.0f*(w0*w1*Exy + w0*w2*Exz + w1*w2*Eyz);
  const float var  = ez2 - mean*mean;
  const float sc   = gamma[o] * rsqrtf(var + EPS_);
  s_out[o] = sc;
  t_out[o] = beta[o] - mean * sc;
}

// ---------------------------------------------------------------------------
// WMMA GEMM: Z(rows x 128) = X(rows x KDIM) * W(128 x KDIM)^T  (f16 in, f32 acc)
// One 16x16 tile per wave; 8 waves/block cover all 128 output channels.
// Also accumulates per-channel sum / sumsq for training-mode BN.
// A/B register packing follows CDNA5 ISA 7.12.2 16-bit layouts.
// ---------------------------------------------------------------------------
template<int KDIM>
__global__ __launch_bounds__(256)
void wmma_gemm_stats_kernel(const float* __restrict__ X, const float* __restrict__ W,
                            float* __restrict__ Z, float* __restrict__ stats, int rows) {
  __shared__ float ssum[CO], ssq[CO];
  const int t = threadIdx.x;
  if (t < CO) { ssum[t] = 0.0f; ssq[t] = 0.0f; }
  __syncthreads();

  const int lane = t & 31;
  const int wave = t >> 5;
  const int half = lane >> 4;
  const int lid  = lane & 15;
  const int mo   = blockIdx.x * 16;  // row tile
  const int oc   = wave * 16;        // channel tile
  (void)rows;

  const float* Xr = X + (size_t)(mo + lid) * KDIM;
  const float* Wr = W + (size_t)(oc + lid) * KDIM;

  v8f c = {};
  for (int kb = 0; kb < KDIM; kb += 32) {
    v16h a, bmat;
#pragma unroll
    for (int j = 0; j < 8; ++j) {
      // A (16x32, M x K): VGPR j<4 -> K = half*8+2j ; j>=4 -> K = 16+half*8+2(j-4)
      const int ka = (j < 4) ? (kb + half*8 + 2*j) : (kb + 16 + half*8 + 2*(j-4));
      a[2*j]   = (_Float16)Xr[ka];
      a[2*j+1] = (_Float16)Xr[ka+1];
      // B (32x16, K x N): VGPR j<4 -> K = half*8+2j ; j>=4 -> K = 16+half*8+2(j-4), N = lane&15
      const int kbn = ka;
      bmat[2*j]   = (_Float16)Wr[kbn];
      bmat[2*j+1] = (_Float16)Wr[kbn+1];
    }
    c = __builtin_amdgcn_wmma_f32_16x16x32_f16(false, a, false, bmat,
                                               (short)0, c, false, false);
  }

  float lsum = 0.0f, lsq = 0.0f;
  const int col = oc + lid;
#pragma unroll
  for (int r = 0; r < 8; ++r) {
    const int row = mo + r + 8*half;   // C/D layout: VGPR r -> M=r (lanes 0-15), M=8+r (16-31)
    const float v = c[r];
    Z[(size_t)row * CO + col] = v;
    lsum += v; lsq += v*v;
  }
  atomicAdd(&ssum[col], lsum);
  atomicAdd(&ssq[col], lsq);
  __syncthreads();
  if (t < CO) { atomicAdd(&stats[t], ssum[t]); atomicAdd(&stats[CO + t], ssq[t]); }
}

// ---------------------------------------------------------------------------
// BN finalize: out = [relu]( bn(z) [+ residual] )
// ---------------------------------------------------------------------------
__global__ void bn_finalize_kernel(const float* __restrict__ z, const float* __restrict__ stats,
                                   float inv_cnt, const float* __restrict__ gamma,
                                   const float* __restrict__ beta,
                                   const float* __restrict__ residual, int do_relu,
                                   float* __restrict__ out, int total) {
  const int i = blockIdx.x * blockDim.x + threadIdx.x;
  if (i >= total) return;
  const int o = i & (CO - 1);
  const float mean = stats[o] * inv_cnt;
  const float var  = stats[CO + o] * inv_cnt - mean*mean;
  float v = (z[i] - mean) * rsqrtf(var + EPS_) * gamma[o] + beta[o];
  if (residual) v += residual[i];
  if (do_relu)  v = fmaxf(v, 0.0f);
  out[i] = v;
}

// ---------------------------------------------------------------------------
// Fused gather + positional branch + max over K:
//   out[b,m,o] = max_k( src[b, sel[k], o] + relu( sc_o*(W3_o . gp_k) + bi_o ) )
// One wave per (b,m), 4 channels per lane.
// ---------------------------------------------------------------------------
__global__ __launch_bounds__(256)
void gather_max_kernel(const float* __restrict__ src, const int* __restrict__ sel,
                       const float* __restrict__ gp3, const float* __restrict__ W3,
                       const float* __restrict__ sc, const float* __restrict__ bi,
                       float* __restrict__ out, int nsrc) {
  const int wid  = (blockIdx.x * blockDim.x + threadIdx.x) >> 5; // b*M_+m
  const int lane = threadIdx.x & 31;
  const int b = wid / M_;
  const int* S = sel + (size_t)wid * K_;
  const float* G = gp3 + (size_t)wid * K_ * 3;

  float w0[4], w1[4], w2[4], ss[4], tt[4], acc[4];
#pragma unroll
  for (int q = 0; q < 4; ++q) {
    const int o = lane + 32*q;
    w0[q] = W3[o*3+0]; w1[q] = W3[o*3+1]; w2[q] = W3[o*3+2];
    ss[q] = sc[o]; tt[q] = bi[o];
    acc[q] = -3.0e38f;
  }
  for (int k = 0; k < K_; ++k) {
    const int idx = S[k];
    const float* row = src + ((size_t)b * nsrc + idx) * CO;
    if (k + 1 < K_)
      __builtin_prefetch(src + ((size_t)b * nsrc + S[k+1]) * CO, 0, 1);
    const float gx = G[k*3+0], gy = G[k*3+1], gz = G[k*3+2];
#pragma unroll
    for (int q = 0; q < 4; ++q) {
      const int o = lane + 32*q;
      const float zv = w0[q]*gx + w1[q]*gy + w2[q]*gz;
      const float pe = fmaxf(zv * ss[q] + tt[q], 0.0f);
      acc[q] = fmaxf(acc[q], row[o] + pe);
    }
  }
  float* O = out + (size_t)wid * CO;
#pragma unroll
  for (int q = 0; q < 4; ++q) O[lane + 32*q] = acc[q];
}

// ---------------------------------------------------------------------------
// Launcher
// ---------------------------------------------------------------------------
extern "C" void kernel_launch(void* const* d_in, const int* in_sizes, int n_in,
                              void* d_out, int out_size, void* d_ws, size_t ws_size,
                              hipStream_t stream) {
  (void)in_sizes; (void)n_in; (void)out_size; (void)ws_size;
  const float* pos  = (const float*)d_in[0];
  const float* x    = (const float*)d_in[1];
  const float* W_x  = (const float*)d_in[2];
  const float* g_x  = (const float*)d_in[3];
  const float* b_x  = (const float*)d_in[4];
  const float* W_ps = (const float*)d_in[5];
  const float* g_ps = (const float*)d_in[6];
  const float* b_ps = (const float*)d_in[7];
  const float* W_pe = (const float*)d_in[8];
  const float* g_pe = (const float*)d_in[9];
  const float* b_pe = (const float*)d_in[10];
  const float* W1   = (const float*)d_in[11];
  const float* g1   = (const float*)d_in[12];
  const float* b1   = (const float*)d_in[13];
  const float* W2a  = (const float*)d_in[14];
  const float* g2a  = (const float*)d_in[15];
  const float* b2a  = (const float*)d_in[16];
  const float* W2b  = (const float*)d_in[17];
  const float* g2b  = (const float*)d_in[18];
  const float* b2b  = (const float*)d_in[19];

  float* out       = (float*)d_out;
  float* centroids = out;                       // (B,M,3)
  float* xf_out    = out + (size_t)B_*M_*3;     // (B,M,128)

  // workspace layout (floats)
  float* ws     = (float*)d_ws;
  float* zbuf   = ws;                                  // 32768*128
  float* feats  = zbuf   + (size_t)B_*N_*CO;           // 32768*128
  float* featsi = feats  + (size_t)B_*N_*CO;           // 8192*128
  float* gbuf   = featsi + (size_t)B_*M_*CO;           // 8192*128
  float* hbuf   = gbuf   + (size_t)B_*M_*CO;           // 8192*128
  float* xfbuf  = hbuf   + (size_t)B_*M_*CO;           // 8192*128
  float* gpb    = xfbuf  + (size_t)B_*M_*CO;           // B*M*K*3
  float* gpeb   = gpb    + (size_t)B_*M_*K_*3;         // B*M*K*3
  float* stats  = gpeb   + (size_t)B_*M_*K_*3;
  float* statsX   = stats;              // 256
  float* statsBlk = stats + 256;        // 6 * 256  (W1,W2a,W2b per block)
  float* statsPS  = stats + 1792;       // 16
  float* statsPE  = stats + 1808;       // 16
  float* sPS      = stats + 1824;       // 128
  float* tPS      = sPS + CO;
  float* sPE      = tPS + CO;
  float* tPE      = sPE + CO;
  int*   sel1     = (int*)(stats + 2560);
  int*   sel2     = sel1 + (size_t)B_*M_*K_;

  const float invN  = 1.0f / (float)(B_*N_);
  const float invM  = 1.0f / (float)(B_*M_);
  const float invMK = 1.0f / (float)(B_*M_*K_);

  // 0) zero all BN accumulators (7*256 GEMM stats + 2*16 gp stats = 1824)
  zero_kernel<<<8, 256, 0, stream>>>(stats, 1824);

  // 1) FPS -> centroids (first tuple output)
  fps_kernel<<<B_, 1024, 0, stream>>>(pos, centroids);

  // 2) ball queries (one wave per centroid; B*M waves -> 1024 blocks of 256)
  ballq_kernel<<<1024, 256, 0, stream>>>(pos,       centroids, sel1, N_, 0.1f*0.1f);
  ballq_kernel<<<1024, 256, 0, stream>>>(centroids, centroids, sel2, M_, 0.2f*0.2f);

  // 3) normalized offsets + moments for analytic BN folding of 3->128 convs
  gp_stats_kernel<<<(B_*M_*K_)/256, 256, 0, stream>>>(pos,       centroids, sel1, gpb,  statsPS, N_, 1.0f/0.1f);
  gp_stats_kernel<<<(B_*M_*K_)/256, 256, 0, stream>>>(centroids, centroids, sel2, gpeb, statsPE, M_, 1.0f/0.2f);
  bn_fold3_kernel<<<1, CO, 0, stream>>>(W_ps, statsPS, invMK, g_ps, b_ps, sPS, tPS);
  bn_fold3_kernel<<<1, CO, 0, stream>>>(W_pe, statsPE, invMK, g_pe, b_pe, sPE, tPE);

  // 4) x_mlp: WMMA GEMM (32768 x 64 x 128) + BN stats, then BN+ReLU -> feats
  wmma_gemm_stats_kernel<CIN><<<(B_*N_)/16, 256, 0, stream>>>(x, W_x, zbuf, statsX, B_*N_);
  bn_finalize_kernel<<<(B_*N_*CO)/256, 256, 0, stream>>>(zbuf, statsX, invN, g_x, b_x,
                                                         nullptr, 1, feats, B_*N_*CO);

  // 5) set abstraction: xf = max_k( feats[sel1] + relu(fold(W_ps . gp)) )
  gather_max_kernel<<<1024, 256, 0, stream>>>(feats, sel1, gpb, W_ps, sPS, tPS, xfbuf, N_);

  // 6) InvResMLP blocks
  const float* xf_src = xfbuf;
  for (int i = 0; i < L_; ++i) {
    float* stA = statsBlk + (size_t)(3*i + 0) * 256;
    float* stB = statsBlk + (size_t)(3*i + 1) * 256;
    float* stC = statsBlk + (size_t)(3*i + 2) * 256;

    // feats_i = relu(bn(W1 . xf))
    wmma_gemm_stats_kernel<CO><<<(B_*M_)/16, 256, 0, stream>>>(xf_src, W1 + (size_t)i*CO*CO,
                                                               zbuf, stA, B_*M_);
    bn_finalize_kernel<<<(B_*M_*CO)/256, 256, 0, stream>>>(zbuf, stA, invM,
                                                           g1 + (size_t)i*CO, b1 + (size_t)i*CO,
                                                           nullptr, 1, featsi, B_*M_*CO);
    // g = max_k( feats_i[sel2] + pe )   (pe recomputed on the fly from gpe)
    gather_max_kernel<<<1024, 256, 0, stream>>>(featsi, sel2, gpeb, W_pe, sPE, tPE, gbuf, M_);

    // h = relu(bn(W2a . g))
    wmma_gemm_stats_kernel<CO><<<(B_*M_)/16, 256, 0, stream>>>(gbuf, W2a + (size_t)i*CO*CO,
                                                               zbuf, stB, B_*M_);
    bn_finalize_kernel<<<(B_*M_*CO)/256, 256, 0, stream>>>(zbuf, stB, invM,
                                                           g2a + (size_t)i*CO, b2a + (size_t)i*CO,
                                                           nullptr, 1, hbuf, B_*M_*CO);
    // xf = relu(feats_i + bn(W2b . h))
    float* dst = (i == L_ - 1) ? xf_out : xfbuf;
    wmma_gemm_stats_kernel<CO><<<(B_*M_)/16, 256, 0, stream>>>(hbuf, W2b + (size_t)i*CO*CO,
                                                               zbuf, stC, B_*M_);
    bn_finalize_kernel<<<(B_*M_*CO)/256, 256, 0, stream>>>(zbuf, stC, invM,
                                                           g2b + (size_t)i*CO, b2b + (size_t)i*CO,
                                                           featsi, 1, dst, B_*M_*CO);
    xf_src = dst;
  }
}
